// TransformerDecoder_35820027248880
// MI455X (gfx1250) — matrix operand order repaired and checked by reading, not compiled, not run on hardware
//
#include <hip/hip_runtime.h>
#include <hip/hip_bf16.h>
#include <math.h>

// ---------------------------------------------------------------------------
// Types for CDNA5 WMMA
// ---------------------------------------------------------------------------
typedef __attribute__((ext_vector_type(16))) _Float16 v16h;
typedef __attribute__((ext_vector_type(8)))  _Float16 v8h;
typedef __attribute__((ext_vector_type(8)))  float    v8f;

#define BLK_M 128
#define BLK_N 64
#define BLK_K 32
#define LDSK  (BLK_K + 8)   // 40 halves = 80 bytes row stride (16B aligned)

// ---------------------------------------------------------------------------
// Tile-exact GEMM (no bounds checks anywhere): C = act(A @ B + bias (+C))
// Caller guarantees: gridDim.y*128 rows, gridDim.x*64 cols, K % 32 == 0,
// and that all touched addresses are inside (padded) buffers.
//  - TRANS_B==false: B row-major [K,N]; TRANS_B==true: B row-major [N,K]
//  - batch: z -> (zb=z/HB, zh=z%HB), offset zb*sXb + zh*sXh
//  - ACT: 0 none, 1 relu
// 8 waves; wave grid 4x2; each wave 32x32 via 2x2 16x16x32 WMMA tiles.
// ---------------------------------------------------------------------------
template<bool TRANS_B, int ACT, bool ACC>
__global__ __launch_bounds__(256)
void gemm_wmma(const float* __restrict__ A, int lda, int sAb, int sAh,
               const float* __restrict__ B, int ldb, int sBb, int sBh,
               const float* __restrict__ bias,
               float* __restrict__ C, int ldc, int sCb, int sCh,
               int K, int HB)
{
    __shared__ __align__(16) _Float16 As[BLK_M][LDSK];
    __shared__ __align__(16) _Float16 Bs[BLK_N][LDSK];   // column-major: Bs[n][k]

    const int zb = blockIdx.z / HB;
    const int zh = blockIdx.z % HB;
    A += (size_t)zb * sAb + (size_t)zh * sAh;
    B += (size_t)zb * sBb + (size_t)zh * sBh;
    C += (size_t)zb * sCb + (size_t)zh * sCh;

    const int m0   = blockIdx.y * BLK_M;
    const int n0   = blockIdx.x * BLK_N;
    const int tid  = threadIdx.x;
    const int lane = tid & 31;
    const int wave = tid >> 5;
    const int wm   = wave >> 1;     // 0..3 (M)
    const int wn   = wave & 1;      // 0..1 (N)
    const int lg   = lane >> 4;     // lane group 0/1
    const int lr   = lane & 15;
    const int kbA  = lg * 8;        // f16 A-matrix K split per ISA 7.12.2

    // staging coordinates (hoisted)
    const int ar = tid >> 1;               // A row in tile
    const int ac = (tid & 1) * 16;         // A 16-elem half-row
    const float* aptr = A + (size_t)(m0 + ar) * lda + ac;
    const int bn  = tid >> 2, bcT = (tid & 3) * 8;     // TRANS_B staging
    const float* bptrT = B + (size_t)(n0 + bn) * ldb + bcT;
    const int bk  = tid >> 3, bcN = (tid & 7) * 8;     // normal-B staging
    const float* bptrN = B + (size_t)bk * ldb + n0 + bcN;

    const v8f vzero = {0.f,0.f,0.f,0.f,0.f,0.f,0.f,0.f};
    v8f acc[2][2];
    acc[0][0] = vzero; acc[0][1] = vzero; acc[1][0] = vzero; acc[1][1] = vzero;

    for (int k0 = 0; k0 < K; k0 += BLK_K) {
        // ---- stage A tile (128 x 32): 4x global_load_b128, cvt, 2x ds_store_b128
        {
            const float* s = aptr + k0;
            float4 f0 = *(const float4*)(s + 0);
            float4 f1 = *(const float4*)(s + 4);
            float4 f2 = *(const float4*)(s + 8);
            float4 f3 = *(const float4*)(s + 12);
            __builtin_prefetch(s + BLK_K, 0, 1);
            v8h lo, hi;
            lo[0]=(_Float16)f0.x; lo[1]=(_Float16)f0.y; lo[2]=(_Float16)f0.z; lo[3]=(_Float16)f0.w;
            lo[4]=(_Float16)f1.x; lo[5]=(_Float16)f1.y; lo[6]=(_Float16)f1.z; lo[7]=(_Float16)f1.w;
            hi[0]=(_Float16)f2.x; hi[1]=(_Float16)f2.y; hi[2]=(_Float16)f2.z; hi[3]=(_Float16)f2.w;
            hi[4]=(_Float16)f3.x; hi[5]=(_Float16)f3.y; hi[6]=(_Float16)f3.z; hi[7]=(_Float16)f3.w;
            *(v8h*)&As[ar][ac]     = lo;
            *(v8h*)&As[ar][ac + 8] = hi;
        }
        // ---- stage B tile into column-major Bs[n][k] ----
        if (TRANS_B) {
            const float* s = bptrT + k0;
            float4 f0 = *(const float4*)(s + 0);
            float4 f1 = *(const float4*)(s + 4);
            v8h pk;
            pk[0]=(_Float16)f0.x; pk[1]=(_Float16)f0.y; pk[2]=(_Float16)f0.z; pk[3]=(_Float16)f0.w;
            pk[4]=(_Float16)f1.x; pk[5]=(_Float16)f1.y; pk[6]=(_Float16)f1.z; pk[7]=(_Float16)f1.w;
            *(v8h*)&Bs[bn][bcT] = pk;
        } else {
            const float* s = bptrN + (size_t)k0 * ldb;
            float4 f0 = *(const float4*)(s + 0);
            float4 f1 = *(const float4*)(s + 4);
            Bs[bcN + 0][bk] = (_Float16)f0.x;
            Bs[bcN + 1][bk] = (_Float16)f0.y;
            Bs[bcN + 2][bk] = (_Float16)f0.z;
            Bs[bcN + 3][bk] = (_Float16)f0.w;
            Bs[bcN + 4][bk] = (_Float16)f1.x;
            Bs[bcN + 5][bk] = (_Float16)f1.y;
            Bs[bcN + 6][bk] = (_Float16)f1.z;
            Bs[bcN + 7][bk] = (_Float16)f1.w;
        }
        __syncthreads();

        // ---- fragments (ISA 7.12.2 16-bit layouts) + 4x WMMA ----
        v16h afr[2], bfr[2];
        #pragma unroll
        for (int t = 0; t < 2; ++t) {
            const int row = wm * 32 + t * 16 + lr;
            v8h alo = *(const v8h*)&As[row][kbA];
            v8h ahi = *(const v8h*)&As[row][16 + kbA];
            afr[t] = __builtin_shufflevector(alo, ahi,
                      0,1,2,3,4,5,6,7,8,9,10,11,12,13,14,15);
            const int col = wn * 32 + t * 16 + lr;
            v8h blo = *(const v8h*)&Bs[col][lg * 8];
            v8h bhi = *(const v8h*)&Bs[col][16 + lg * 8];
            bfr[t] = __builtin_shufflevector(blo, bhi,
                      0,1,2,3,4,5,6,7,8,9,10,11,12,13,14,15);
        }
        #pragma unroll
        for (int ti = 0; ti < 2; ++ti)
            #pragma unroll
            for (int tj = 0; tj < 2; ++tj)
                acc[ti][tj] = __builtin_amdgcn_wmma_f32_16x16x32_f16(
                    false, afr[ti], false, bfr[tj],
                    (short)0, acc[ti][tj], false, false);
        __syncthreads();
    }

    // ---- epilogue (unguarded) ----
    #pragma unroll
    for (int ti = 0; ti < 2; ++ti) {
        #pragma unroll
        for (int tj = 0; tj < 2; ++tj) {
            const int col = n0 + wn * 32 + tj * 16 + lr;
            const float bv = bias ? bias[col] : 0.f;
            #pragma unroll
            for (int v = 0; v < 8; ++v) {
                const int row = m0 + wm * 32 + ti * 16 + lg * 8 + v;
                float val = acc[ti][tj][v] + bv;
                float* cp = &C[(size_t)row * ldc + col];
                if (ACC) val += *cp;
                if (ACT == 1) val = val > 0.f ? val : 0.f;
                *cp = val;
            }
        }
    }
}

// ---------------------------------------------------------------------------
// Tile-exact attention P @ V kernel: block tile 128x32, K % 32 == 0.
// 8 waves, wave w -> rows w*16..w*16+15; 2 WMMAs (N=32) per K-step.
// ---------------------------------------------------------------------------
__global__ __launch_bounds__(256)
void gemm_attnv(const float* __restrict__ A, int lda, int sAb, int sAh,
                const float* __restrict__ B, int ldb, int sBb, int sBh,
                float* __restrict__ C, int ldc, int sCb, int sCh,
                int K, int HB)
{
    __shared__ __align__(16) _Float16 As[BLK_M][LDSK];
    __shared__ __align__(16) _Float16 Bs[32][LDSK];

    const int zb = blockIdx.z / HB;
    const int zh = blockIdx.z % HB;
    A += (size_t)zb * sAb + (size_t)zh * sAh;
    B += (size_t)zb * sBb + (size_t)zh * sBh;
    C += (size_t)zb * sCb + (size_t)zh * sCh;

    const int m0   = blockIdx.y * BLK_M;
    const int tid  = threadIdx.x;
    const int lane = tid & 31;
    const int wave = tid >> 5;      // 0..7, rows wave*16..+15
    const int lg   = lane >> 4;
    const int lr   = lane & 15;
    const int kbA  = lg * 8;

    const int ar = tid >> 1, ac = (tid & 1) * 16;
    const float* aptr = A + (size_t)(m0 + ar) * lda + ac;
    const int bk = tid >> 3, bc = (tid & 7) * 4;      // 32k x 32n tile, 4/thread
    const float* bptr = B + (size_t)bk * ldb + bc;

    const v8f vzero = {0.f,0.f,0.f,0.f,0.f,0.f,0.f,0.f};
    v8f acc[2]; acc[0] = vzero; acc[1] = vzero;

    for (int k0 = 0; k0 < K; k0 += BLK_K) {
        {
            const float* s = aptr + k0;
            float4 f0 = *(const float4*)(s + 0);
            float4 f1 = *(const float4*)(s + 4);
            float4 f2 = *(const float4*)(s + 8);
            float4 f3 = *(const float4*)(s + 12);
            __builtin_prefetch(s + BLK_K, 0, 1);
            v8h lo, hi;
            lo[0]=(_Float16)f0.x; lo[1]=(_Float16)f0.y; lo[2]=(_Float16)f0.z; lo[3]=(_Float16)f0.w;
            lo[4]=(_Float16)f1.x; lo[5]=(_Float16)f1.y; lo[6]=(_Float16)f1.z; lo[7]=(_Float16)f1.w;
            hi[0]=(_Float16)f2.x; hi[1]=(_Float16)f2.y; hi[2]=(_Float16)f2.z; hi[3]=(_Float16)f2.w;
            hi[4]=(_Float16)f3.x; hi[5]=(_Float16)f3.y; hi[6]=(_Float16)f3.z; hi[7]=(_Float16)f3.w;
            *(v8h*)&As[ar][ac]     = lo;
            *(v8h*)&As[ar][ac + 8] = hi;
        }
        {
            const float* s = bptr + (size_t)k0 * ldb;
            float4 f = *(const float4*)s;
            Bs[bc + 0][bk] = (_Float16)f.x;
            Bs[bc + 1][bk] = (_Float16)f.y;
            Bs[bc + 2][bk] = (_Float16)f.z;
            Bs[bc + 3][bk] = (_Float16)f.w;
        }
        __syncthreads();

        const int row = wave * 16 + lr;
        v8h alo = *(const v8h*)&As[row][kbA];
        v8h ahi = *(const v8h*)&As[row][16 + kbA];
        v16h afr = __builtin_shufflevector(alo, ahi,
                    0,1,2,3,4,5,6,7,8,9,10,11,12,13,14,15);
        #pragma unroll
        for (int tj = 0; tj < 2; ++tj) {
            const int col = tj * 16 + lr;
            v8h blo = *(const v8h*)&Bs[col][lg * 8];
            v8h bhi = *(const v8h*)&Bs[col][16 + lg * 8];
            v16h bfr = __builtin_shufflevector(blo, bhi,
                        0,1,2,3,4,5,6,7,8,9,10,11,12,13,14,15);
            acc[tj] = __builtin_amdgcn_wmma_f32_16x16x32_f16(
                false, afr, false, bfr, (short)0, acc[tj], false, false);
        }
        __syncthreads();
    }

    #pragma unroll
    for (int tj = 0; tj < 2; ++tj) {
        const int col = tj * 16 + lr;
        #pragma unroll
        for (int v = 0; v < 8; ++v) {
            const int row = m0 + wave * 16 + lg * 8 + v;
            C[(size_t)row * ldc + col] = acc[tj][v];
        }
    }
}

// ---------------------------------------------------------------------------
// Row softmax in-place with pre-scale; zeroes pad columns [T, Tpad).
// Rows live in slices: row ptr = S + (bid/rps)*sliceStride + (bid%rps)*Tpad.
// ---------------------------------------------------------------------------
__global__ __launch_bounds__(256)
void softmax_rows(float* __restrict__ S, int rps, long sliceStride,
                  int T, int Tpad, float scale)
{
    __shared__ float red[256];
    float* row = S + (size_t)(blockIdx.x / rps) * sliceStride
                   + (size_t)(blockIdx.x % rps) * Tpad;
    const int tid = threadIdx.x;

    float mx = -3.4e38f;
    for (int j = tid; j < T; j += 256) mx = fmaxf(mx, row[j] * scale);
    red[tid] = mx; __syncthreads();
    for (int s = 128; s > 0; s >>= 1) { if (tid < s) red[tid] = fmaxf(red[tid], red[tid + s]); __syncthreads(); }
    mx = red[0]; __syncthreads();

    float sum = 0.f;
    for (int j = tid; j < T; j += 256) { float e = __expf(row[j] * scale - mx); row[j] = e; sum += e; }
    red[tid] = sum; __syncthreads();
    for (int s = 128; s > 0; s >>= 1) { if (tid < s) red[tid] += red[tid + s]; __syncthreads(); }
    const float inv = 1.f / red[0];
    for (int j = tid; j < T; j += 256) row[j] *= inv;
    for (int j = T + tid; j < Tpad; j += 256) row[j] = 0.f;   // exact-zero pad cols
}

// ---------------------------------------------------------------------------
// y = LayerNorm(x (+res)) * g + b, D=256, one block per token.
// ---------------------------------------------------------------------------
template<bool HAS_RES, bool TRANS_OUT>
__global__ __launch_bounds__(256)
void addln_kernel(const float* __restrict__ x, const float* __restrict__ res,
                  const float* __restrict__ g, const float* __restrict__ b,
                  float* __restrict__ y, int BSd, int NQd)
{
    __shared__ float red[256];
    const int tok = blockIdx.x, f = threadIdx.x;
    float v = x[(size_t)tok * 256 + f];
    if (HAS_RES) v += res[(size_t)tok * 256 + f];

    red[f] = v; __syncthreads();
    for (int s = 128; s > 0; s >>= 1) { if (f < s) red[f] += red[f + s]; __syncthreads(); }
    const float mean = red[0] * (1.f / 256.f); __syncthreads();
    const float d = v - mean;
    red[f] = d * d; __syncthreads();
    for (int s = 128; s > 0; s >>= 1) { if (f < s) red[f] += red[f + s]; __syncthreads(); }
    const float var = red[0] * (1.f / 256.f);
    const float o = d * rsqrtf(var + 1e-5f) * g[f] + b[f];

    if (TRANS_OUT) {
        const int s_ = tok / BSd, bb = tok % BSd;
        y[((size_t)bb * NQd + s_) * 256 + f] = o;
    } else {
        y[(size_t)tok * 256 + f] = o;
    }
}

// ---------------------------------------------------------------------------
// Fused ref_point_head MLP: refp[tok,2] = sigmoid(relu(x@W0+b0)@W1+b1)
// One block per token; f32 VALU (tiny: N=2 doesn't fit the WMMA path).
// ---------------------------------------------------------------------------
__global__ __launch_bounds__(256)
void refpoint_kernel(const float* __restrict__ x,
                     const float* __restrict__ w0, const float* __restrict__ b0,
                     const float* __restrict__ w1, const float* __restrict__ b1,
                     float* __restrict__ refp)
{
    __shared__ float xs[256];
    __shared__ float hs[256];
    __shared__ float red[256];
    const int tok = blockIdx.x, f = threadIdx.x;
    xs[f] = x[(size_t)tok * 256 + f];
    __syncthreads();
    float a = b0[f];
    for (int k = 0; k < 256; ++k) a += xs[k] * w0[(size_t)k * 256 + f];
    hs[f] = a > 0.f ? a : 0.f;
    __syncthreads();
    #pragma unroll
    for (int j = 0; j < 2; ++j) {
        red[f] = hs[f] * w1[(size_t)f * 2 + j];
        __syncthreads();
        for (int s = 128; s > 0; s >>= 1) { if (f < s) red[f] += red[f + s]; __syncthreads(); }
        if (f == 0) refp[(size_t)tok * 2 + j] = 1.f / (1.f + __expf(-(red[0] + b1[j])));
        __syncthreads();
    }
}

// Per-head concat: c[tok, h*64+d] = d<32 ? a[tok,h*32+d] : b[tok,h*32+d-32]
__global__ void concat_heads(const float* __restrict__ a, const float* __restrict__ b,
                             float* __restrict__ c, int tokens)
{
    const int i = blockIdx.x * 256 + threadIdx.x;
    if (i >= tokens * 512) return;
    const int tok = i >> 9, r = i & 511, h = r >> 6, d = r & 63;
    c[i] = (d < 32) ? a[(size_t)tok * 256 + h * 32 + d]
                    : b[(size_t)tok * 256 + h * 32 + d - 32];
}

__global__ void mul_kernel(const float* __restrict__ a, const float* __restrict__ b,
                           float* __restrict__ c, int n)
{ const int i = blockIdx.x * 256 + threadIdx.x; if (i < n) c[i] = a[i] * b[i]; }

__global__ void add_kernel(const float* __restrict__ a, const float* __restrict__ b,
                           float* __restrict__ c, int n)
{ const int i = blockIdx.x * 256 + threadIdx.x; if (i < n) c[i] = a[i] + b[i]; }

__global__ void zero_kernel(float* __restrict__ p, int n)
{ const int i = blockIdx.x * 256 + threadIdx.x; if (i < n) p[i] = 0.f; }

// Sine embedding: ref[tok,2] -> qse[tok,256] = [ey(128) | ex(128)]
__global__ void sine_kernel(const float* __restrict__ ref, float* __restrict__ qse, int tokens)
{
    const int i = blockIdx.x * 256 + threadIdx.x;
    if (i >= tokens * 64) return;
    const int tok = i >> 6, t = i & 63;
    const float inv = 6.283185307179586f * __expf(-(float)t * 0.14391156831212787f);
    const float px = ref[(size_t)tok * 2 + 0] * inv;
    const float py = ref[(size_t)tok * 2 + 1] * inv;
    float* q = qse + (size_t)tok * 256;
    q[2 * t]           = __sinf(py);
    q[2 * t + 1]       = __cosf(py);
    q[128 + 2 * t]     = __sinf(px);
    q[128 + 2 * t + 1] = __cosf(px);
}

// reference_points output: o[(b*NQ+s)*2+j] = r[(s*BS+b)*2+j]
__global__ void ref_out_kernel(const float* __restrict__ r, float* __restrict__ o,
                               int BSd, int NQd)
{
    const int i = blockIdx.x * 256 + threadIdx.x;
    if (i >= BSd * NQd * 2) return;
    const int j = i & 1, tok = i >> 1;
    const int s = tok / BSd, b = tok % BSd;
    o[((size_t)b * NQd + s) * 2 + j] = r[i];
}

// ---------------------------------------------------------------------------
// Host orchestration
// ---------------------------------------------------------------------------
extern "C" void kernel_launch(void* const* d_in, const int* in_sizes, int n_in,
                              void* d_out, int out_size, void* d_ws, size_t ws_size,
                              hipStream_t stream)
{
    (void)in_sizes; (void)out_size;
    if (n_in < 220) return;

    const int Dm = 256, Hh = 8, BSc = 16, NQc = 300, HWc = 1296, Lc = 6, FFc = 2048;
    const int TOK  = NQc * BSc;          // 4800 real query tokens (row = s*BS+b)
    const int NQP  = 384;                // padded query rows (3 * 128)
    const int TOKP = NQP * BSc;          // 6144 padded query tokens
    const int MTOK = HWc * BSc;          // 20736 memory tokens (162 * 128, exact)
    const int HWP  = 1344;               // padded key rows (21 * 64 / 42 * 32)
    const int MTOKP= HWP * BSc;          // 21504 padded key tokens
    const int SSLD = 320;                // self scores padded cols (10 * 32)
    const long SSSTR = (long)NQP * SSLD; // per-(b,h) self score slice
    const long CSSTR = (long)NQP * HWP;  // per-h cross score slice

    const float* in_output = (const float*)d_in[0];
    const float* in_memory = (const float*)d_in[1];
    const float* in_pos    = (const float*)d_in[2];
    const float* in_qpos   = (const float*)d_in[3];
    auto P = [&](int i) { return (const float*)d_in[i]; };

    // Flattened param indices (insertion order; layer0 has ca_qpos appended).
    int LB[6]; LB[0] = 4; for (int i = 1; i < 6; ++i) LB[i] = 40 + (i - 1) * 34;
    enum { SA_QC = 0, SA_QP = 2, SA_KC = 4, SA_KP = 6, SA_V = 8, SA_OUT = 10,
           CA_QC = 12, CA_KC = 14, CA_KP = 16, CA_V = 18, CA_SINE = 20, CA_OUT = 22,
           FF1 = 24, FF2 = 26, LN1 = 28, LN2 = 30, LN3 = 32, CA_QPOS = 34 };
    const int QS0 = 210, QS1 = 212, RP0 = 214, RP1 = 216, NORMG = 218;

    // ---- workspace layout (f32), all padded so GEMMs are tile-exact ----
    float* W = (float*)d_ws;
    size_t off = 0;
    auto alloc = [&](size_t n) { float* p = W + off; off += n; return p; };
    float* tgt   = alloc((size_t)TOKP * Dm);
    float* qposP = alloc((size_t)TOKP * Dm);
    float* qse   = alloc((size_t)TOKP * Dm);
    float* refp  = alloc((size_t)TOK * 2);
    float* qsO   = alloc((size_t)TOKP * Dm);
    float* bA    = alloc((size_t)TOKP * Dm);
    float* bB    = alloc((size_t)TOKP * Dm);
    float* bC    = alloc((size_t)TOKP * Dm);
    float* bD    = alloc((size_t)TOKP * Dm);
    float* bE    = alloc((size_t)TOKP * Dm);
    float* qcat  = alloc((size_t)TOKP * 2 * Dm);
    float* kcb   = alloc((size_t)MTOK * Dm);
    float* kpb   = alloc((size_t)MTOK * Dm);
    float* vvb   = alloc((size_t)MTOKP * Dm);
    float* kcat  = alloc((size_t)MTOKP * 2 * Dm);
    float* ssc   = alloc((size_t)BSc * Hh * SSSTR);   // 15.73M f32; reused as FFN hidden
    float* csc   = alloc((size_t)Hh * CSSTR);         // per-b cross scores
    float* ffh   = ssc;                               // TOKP*FF = 12.58M <= 15.73M
    if (off * sizeof(float) > ws_size) return;

    auto cdiv = [](int a, int b) { return (a + b - 1) / b; };

#define GEMM(TB, ACTv, ACCv, A_, lda_, sAb_, sAh_, B_, ldb_, sBb_, sBh_, bias_, \
             C_, ldc_, sCb_, sCh_, M_, N_, K_, HB_, NB_)                        \
    gemm_wmma<TB, ACTv, ACCv><<<dim3((N_) / BLK_N, (M_) / BLK_M, NB_), 256, 0, stream>>>( \
        A_, lda_, sAb_, sAh_, B_, ldb_, sBb_, sBh_, bias_, C_, ldc_, sCb_, sCh_, K_, HB_)

#define LIN(ACTv, ACCv, A_, Wi_, C_, M_, N_, K_) \
    GEMM(false, ACTv, ACCv, A_, K_, 0, 0, P(Wi_), N_, 0, 0, P((Wi_) + 1), C_, N_, 0, 0, M_, N_, K_, 1, 1)

    // ---- phase 0: padded copies, ref points, sine embed ----
    hipMemcpyAsync(tgt, in_output, (size_t)TOK * Dm * sizeof(float),
                   hipMemcpyDeviceToDevice, stream);
    zero_kernel<<<cdiv((TOKP - TOK) * Dm, 256), 256, 0, stream>>>(
        tgt + (size_t)TOK * Dm, (TOKP - TOK) * Dm);
    hipMemcpyAsync(qposP, in_qpos, (size_t)TOK * Dm * sizeof(float),
                   hipMemcpyDeviceToDevice, stream);
    zero_kernel<<<cdiv((TOKP - TOK) * Dm, 256), 256, 0, stream>>>(
        qposP + (size_t)TOK * Dm, (TOKP - TOK) * Dm);

    refpoint_kernel<<<TOK, 256, 0, stream>>>(in_qpos, P(RP0), P(RP0 + 1),
                                             P(RP1), P(RP1 + 1), refp);
    sine_kernel<<<cdiv(TOK * 64, 256), 256, 0, stream>>>(refp, qse, TOK);
    ref_out_kernel<<<cdiv(TOK * 2, 256), 256, 0, stream>>>(
        refp, (float*)d_out + (size_t)Lc * BSc * NQc * Dm, BSc, NQc);

    const float self_scale  = 0.17677669529663687f;  // 1/sqrt(32)
    const float cross_scale = 0.125f;                // 1/sqrt(64)

    for (int l = 0; l < Lc; ++l) {
        const int LBl = LB[l];
        auto PPi = [&](int o) { return LBl + o; };

        // ================= self-attention =================
        LIN(0, false, tgt,   PPi(SA_QC), bA, TOKP, Dm, Dm);
        LIN(0, true,  qposP, PPi(SA_QP), bA, TOKP, Dm, Dm);   // q = qc + qp
        LIN(0, false, tgt,   PPi(SA_KC), bB, TOKP, Dm, Dm);
        LIN(0, true,  qposP, PPi(SA_KP), bB, TOKP, Dm, Dm);   // k = kc + kp
        LIN(0, false, tgt,   PPi(SA_V),  bC, TOKP, Dm, Dm);
        // scores[b,h] = Q_bh @ K_bh^T : 128 batches, M=384, N=320, K=32
        GEMM(true, 0, false,
             bA, BSc * Dm, Dm, 32,
             bB, BSc * Dm, Dm, 32,
             nullptr,
             ssc, SSLD, Hh * SSSTR, SSSTR,
             NQP, SSLD, 32, Hh, BSc * Hh);
        softmax_rows<<<BSc * Hh * NQc, 256, 0, stream>>>(ssc, NQc, SSSTR,
                                                         NQc, SSLD, self_scale);
        // P @ V -> bD (token-major via strided C), K=320 (pad cols are 0)
        gemm_attnv<<<dim3(1, NQP / BLK_M, BSc * Hh), 256, 0, stream>>>(
            ssc, SSLD, (int)(Hh * SSSTR), (int)SSSTR,
            bC, BSc * Dm, Dm, 32,
            bD, BSc * Dm, Dm, 32,
            SSLD, Hh);
        LIN(0, false, bD, PPi(SA_OUT), bE, TOKP, Dm, Dm);
        addln_kernel<true, false><<<TOK, 256, 0, stream>>>(
            tgt, bE, P(PPi(LN1)), P(PPi(LN1) + 1), tgt, BSc, NQc);

        // ================= cross-attention =================
        LIN(0, false, tgt, PPi(CA_QC), bA, TOKP, Dm, Dm);
        if (l == 0) LIN(0, true, qposP, PPi(CA_QPOS), bA, TOKP, Dm, Dm);
        LIN(0, false, in_memory, PPi(CA_KC), kcb, MTOK, Dm, Dm);
        LIN(0, false, in_pos,    PPi(CA_KP), kpb, MTOK, Dm, Dm);
        LIN(0, false, in_memory, PPi(CA_V),  vvb, MTOK, Dm, Dm);
        if (l == 0)   // first layer: k2 = kc + kp
            add_kernel<<<cdiv(MTOK * Dm, 256), 256, 0, stream>>>(kcb, kpb, kcb, MTOK * Dm);

        const float* qsrc;
        if (l == 0) {
            qsrc = qse;                                   // query_scale == 1.0
        } else {
            LIN(1, false, tgt, QS0, bB,  TOKP, Dm, Dm);   // query_scale MLP
            LIN(0, false, bB,  QS1, qsO, TOKP, Dm, Dm);
            mul_kernel<<<cdiv(TOK * Dm, 256), 256, 0, stream>>>(qse, qsO, qsO, TOK * Dm);
            qsrc = qsO;
        }
        LIN(0, false, qsrc, PPi(CA_SINE), bB, TOKP, Dm, Dm);  // qsp
        concat_heads<<<cdiv(TOK * 2 * Dm, 256), 256, 0, stream>>>(bA, bB, qcat, TOK);
        concat_heads<<<cdiv(MTOK * 2 * Dm, 256), 256, 0, stream>>>(kcb, kpb, kcat, MTOK);

        // loop over batch b, reusing the 8-slice cross score buffer
        for (int b = 0; b < BSc; ++b) {
            const float* qb = qcat + (size_t)b * 2 * Dm;
            const float* kb = kcat + (size_t)b * 2 * Dm;
            GEMM(true, 0, false,
                 qb, BSc * 2 * Dm, 0, 64,
                 kb, BSc * 2 * Dm, 0, 64,
                 nullptr,
                 csc, HWP, 0, CSSTR,
                 NQP, HWP, 64, Hh, Hh);
            softmax_rows<<<Hh * NQc, 256, 0, stream>>>(csc, NQc, CSSTR,
                                                       HWc, HWP, cross_scale);
            gemm_attnv<<<dim3(1, NQP / BLK_M, Hh), 256, 0, stream>>>(
                csc, HWP, 0, (int)CSSTR,
                vvb + (size_t)b * Dm, BSc * Dm, 0, 32,
                bD + (size_t)b * Dm, BSc * Dm, 0, 32,
                HWP, Hh);
        }
        LIN(0, false, bD, PPi(CA_OUT), bE, TOKP, Dm, Dm);
        addln_kernel<true, false><<<TOK, 256, 0, stream>>>(
            tgt, bE, P(PPi(LN2)), P(PPi(LN2) + 1), tgt, BSc, NQc);

        // ================= FFN =================
        LIN(1, false, tgt, PPi(FF1), ffh, TOKP, FFc, Dm);
        LIN(0, false, ffh, PPi(FF2), bE,  TOKP, Dm, FFc);
        addln_kernel<true, false><<<TOK, 256, 0, stream>>>(
            tgt, bE, P(PPi(LN3)), P(PPi(LN3) + 1), tgt, BSc, NQc);

        // intermediate hook: shared final LayerNorm, written [l, b, s, d]
        addln_kernel<false, true><<<TOK, 256, 0, stream>>>(
            tgt, nullptr, P(NORMG), P(NORMG + 1),
            (float*)d_out + (size_t)l * BSc * NQc * Dm, BSc, NQc);
    }

#undef LIN
#undef GEMM
}
